// MaskWeightLayer_76493367542082
// MI455X (gfx1250) — matrix operand order repaired and checked by reading, not compiled, not run on hardware
//
#include <hip/hip_runtime.h>

typedef __attribute__((ext_vector_type(16))) _Float16 v16h;
typedef __attribute__((ext_vector_type(8)))  _Float16 v8h;
typedef __attribute__((ext_vector_type(8)))  float    v8f;
typedef __attribute__((ext_vector_type(4)))  float    v4f;

#define EPSF 1e-8f
#define KDIM 2048
#define MDIM 512
#define NDIM 64

#define TM 16      // rows per GEMM block
#define TC 128     // cols per GEMM block (8 waves x 16)
#define XPITCH 520 // halves; 1040B rows -> 16B aligned, bank-staggered
#define WPITCH 40  // halves; 80B rows -> 16B aligned

static __device__ __forceinline__ v16h cat8(v8h a, v8h b) {
  return __builtin_shufflevector(a, b, 0,1,2,3,4,5,6,7,8,9,10,11,12,13,14,15);
}

// ---------------- K1: thresholds + scores_hard ----------------
__global__ __launch_bounds__(256) void mwl_hard(const float* __restrict__ w_b,
                                                const int* __restrict__ lptr,
                                                float* __restrict__ hard) {
  __shared__ float sthr[NDIM];
  const int tid = threadIdx.x;
  int l = *lptr;
  if (l < 1) l = 1;
  if (l > 16) l = 16;
  if (tid < NDIM) {
    float top[16];
#pragma unroll
    for (int i = 0; i < 16; ++i) top[i] = -3.402823466e38f;
    for (int m = 0; m < MDIM; ++m) {
      float v = w_b[m * NDIM + tid];
      if (v > top[l - 1]) {
        int j = l - 1;
        while (j > 0 && top[j - 1] < v) { top[j] = top[j - 1]; --j; }
        top[j] = v;
      }
    }
    sthr[tid] = top[l - 1];  // l-th largest
  }
  __syncthreads();
  for (int idx = tid; idx < MDIM * NDIM; idx += 256) {
    int n = idx & (NDIM - 1);
    float sh = w_b[idx] - sthr[n] + EPSF;
    sh = fminf(fmaxf(sh, -1.0f), 1.0f);
    hard[idx] = (sh + 1.0f) * 0.5f;
  }
}

// ---------------- K2a: S = x @ w_att via f16 WMMA with hi/lo error compensation ----
__global__ __launch_bounds__(256) void mwl_gemm(const float* __restrict__ x,
                                                const float* __restrict__ w,
                                                float* __restrict__ S) {
  __shared__ _Float16 xh[TM * XPITCH];
  __shared__ _Float16 xl[TM * XPITCH];
  __shared__ _Float16 wth[TC * WPITCH];  // transposed panel: [col][k-chunk 32]
  __shared__ _Float16 wtl[TC * WPITCH];

  const int tid  = threadIdx.x;
  const int lane = tid & 31;
  const int wv   = tid >> 5;
  const int k0   = blockIdx.x * TM;
  const int c0   = blockIdx.y * TC;

  // stage x row-block [16 x 512], split hi/lo f16
  for (int idx = tid; idx < TM * MDIM; idx += 256) {
    int r = idx >> 9, c = idx & (MDIM - 1);
    float v = x[(size_t)(k0 + r) * MDIM + c];
    _Float16 h = (_Float16)v;
    xh[r * XPITCH + c] = h;
    xl[r * XPITCH + c] = (_Float16)(v - (float)h);
  }

  // wave32 WMMA fragment coordinates (ISA 7.12.2, 16-bit A 16x32 / B 32x16)
  const int arow = lane & 15;
  const int kba  = (lane < 16) ? 0 : 8;   // A: halves 0-7 -> K=kba.., 8-15 -> K=kba+16..
  const int bcol = wv * 16 + (lane & 15);
  const int kbb  = (lane < 16) ? 0 : 8;   // B: halves 0-7 -> K=kbb.., 8-15 -> K=kbb+16..

  v8f acc = {};

  for (int it = 0; it < MDIM / 32; ++it) {
    const int I0 = it * 32;
    __syncthreads();
    // stage w panel [32 x 128] transposed, split hi/lo
    for (int idx = tid; idx < 32 * TC; idx += 256) {
      int ki = idx >> 7, c = idx & (TC - 1);
      float v = w[(size_t)(I0 + ki) * MDIM + (c0 + c)];
      _Float16 h = (_Float16)v;
      wth[c * WPITCH + ki] = h;
      wtl[c * WPITCH + ki] = (_Float16)(v - (float)h);
      if (it + 1 < MDIM / 32)
        __builtin_prefetch(&w[(size_t)(I0 + 32 + ki) * MDIM + (c0 + c)], 0, 1);
    }
    __syncthreads();

    const _Float16* xrh = &xh[arow * XPITCH + I0 + kba];
    const _Float16* xrl = &xl[arow * XPITCH + I0 + kba];
    v16h Ah = cat8(*(const v8h*)(xrh), *(const v8h*)(xrh + 16));
    v16h Al = cat8(*(const v8h*)(xrl), *(const v8h*)(xrl + 16));

    const _Float16* brh = &wth[bcol * WPITCH + kbb];
    const _Float16* brl = &wtl[bcol * WPITCH + kbb];
    v16h Bh = cat8(*(const v8h*)(brh), *(const v8h*)(brh + 16));
    v16h Bl = cat8(*(const v8h*)(brl), *(const v8h*)(brl + 16));

    acc = __builtin_amdgcn_wmma_f32_16x16x32_f16(false, Ah, false, Bh, (short)0, acc, false, false);
    acc = __builtin_amdgcn_wmma_f32_16x16x32_f16(false, Al, false, Bh, (short)0, acc, false, false);
    acc = __builtin_amdgcn_wmma_f32_16x16x32_f16(false, Ah, false, Bl, (short)0, acc, false, false);
  }

  // C layout: VGPR j -> row j (lanes 0-15) / j+8 (lanes 16-31), col = lane&15
  const int srow = (lane < 16) ? 0 : 8;
#pragma unroll
  for (int j = 0; j < 8; ++j)
    S[(size_t)(k0 + srow + j) * MDIM + (c0 + bcol)] = acc[j];
}

// ---------------- K2b: row softmax -> scores_soft + EPS ----------------
__global__ __launch_bounds__(256) void mwl_softmax(const float* __restrict__ S,
                                                   float* __restrict__ ss) {
  __shared__ float red[256];
  const int tid = threadIdx.x;
  const int row = blockIdx.x;
  const float* sr = S + (size_t)row * MDIM;
  float v0 = sr[tid], v1 = sr[tid + 256];
  red[tid] = fmaxf(v0, v1);
  __syncthreads();
  for (int s = 128; s > 0; s >>= 1) {
    if (tid < s) red[tid] = fmaxf(red[tid], red[tid + s]);
    __syncthreads();
  }
  float m = red[0];
  __syncthreads();
  float e0 = __expf(v0 - m), e1 = __expf(v1 - m);
  red[tid] = e0 + e1;
  __syncthreads();
  for (int s = 128; s > 0; s >>= 1) {
    if (tid < s) red[tid] += red[tid + s];
    __syncthreads();
  }
  float inv = 1.0f / red[0];
  ss[(size_t)row * MDIM + tid]       = e0 * inv + EPSF;
  ss[(size_t)row * MDIM + tid + 256] = e1 * inv + EPSF;
}

// ---------------- K3: out/mask_weight streaming combine (NT b128 stores) -----
__global__ __launch_bounds__(256) void mwl_combine(const float* __restrict__ x,
                                                   const float* __restrict__ hard,
                                                   const float* __restrict__ ss,
                                                   float* __restrict__ outp,
                                                   float* __restrict__ mwp) {
  const int k = blockIdx.x;
  const int m = blockIdx.y * 16 + (threadIdx.x >> 4);
  const int n = (threadIdx.x & 15) * 4;
  const float s  = ss[(size_t)k * MDIM + m];
  const float xv = x[(size_t)k * MDIM + m];
  v4f h  = *(const v4f*)(hard + m * NDIM + n);
  v4f mw = h * s;
  v4f o  = mw * xv;
  const size_t base = ((size_t)k * MDIM + m) * NDIM + n;
  __builtin_nontemporal_store(mw, (v4f*)(mwp + base));
  __builtin_nontemporal_store(o,  (v4f*)(outp + base));
}

extern "C" void kernel_launch(void* const* d_in, const int* in_sizes, int n_in,
                              void* d_out, int out_size, void* d_ws, size_t ws_size,
                              hipStream_t stream) {
  (void)in_sizes; (void)n_in; (void)out_size; (void)d_ws; (void)ws_size;
  const float* x     = (const float*)d_in[0];  // [2048, 512]
  const float* w_att = (const float*)d_in[1];  // [512, 512]
  const float* w_b   = (const float*)d_in[2];  // [512, 64]
  const int*   lp    = (const int*)d_in[3];    // scalar l

  float* out = (float*)d_out;
  const size_t KMN = (size_t)KDIM * MDIM * NDIM;
  float* hard = out + KMN;                     // [512, 64]
  float* ss   = hard + (size_t)MDIM * NDIM;    // [2048, 512]
  float* mw   = ss + (size_t)KDIM * MDIM;      // [2048, 512, 64]
  float* S    = mw;  // scratch for raw logits; fully overwritten by mwl_combine

  mwl_hard<<<1, 256, 0, stream>>>(w_b, lp, hard);
  mwl_gemm<<<dim3(KDIM / TM, MDIM / TC), 256, 0, stream>>>(x, w_att, S);
  mwl_softmax<<<KDIM, 256, 0, stream>>>(S, ss);
  mwl_combine<<<dim3(KDIM, MDIM / 16), 256, 0, stream>>>(x, hard, ss, out, mw);
}
